// Decoder_77979426226640
// MI455X (gfx1250) — matrix operand order repaired and checked by reading, not compiled, not run on hardware
//
#include <hip/hip_runtime.h>
#include <hip/hip_bf16.h>

// ---------------------------------------------------------------------------
// LSTM decoder for MI455X (gfx1250): bf16 WMMA GEMMs + fp32 state math.
// Per step: map GEMM (256x512x2048) -> lstm0 -> lstm1, 256 steps, all on
// `stream`. Weights converted to bf16 once into d_ws; h ping-pong buffers.
// This round: hardware v_tanh_f32 (branchless tail) + WMMA reuse-A hints.
// ---------------------------------------------------------------------------

typedef __bf16 bf16_t;
typedef __attribute__((ext_vector_type(16))) __bf16 v16bf;
typedef __attribute__((ext_vector_type(8)))  __bf16 v8bf;
typedef __attribute__((ext_vector_type(8)))  float  v8f;

#define HID   512
#define BSZ   256
#define SEQ   256
#define LATD  256
#define FCW   2048   // fc_in width = 4*HID

__device__ __forceinline__ float sigm(float x) { return 1.0f / (1.0f + __expf(-x)); }

// CDNA5 has V_TANH_F32 (TRANS32 op). Use the builtin when the toolchain
// declares it; otherwise a branchless clamp + v_exp_f32 rational form
// (libm tanhf lowers to an EXEC-diverging range split we want to avoid).
__device__ __forceinline__ float fast_tanh(float x) {
#if __has_builtin(__builtin_amdgcn_tanhf)
  return __builtin_amdgcn_tanhf(x);
#else
  float xc = fminf(fmaxf(x, -10.0f), 10.0f);
  float e  = __expf(2.0f * xc);
  return (e - 1.0f) / (e + 1.0f);
#endif
}

// v_wmma_f32_16x16x32_bf16 : D = A(16x32) * B(32x16) + C(16x16)
// RA = hint that this instruction's A operand is cached for the next WMMA.
template <bool RA>
__device__ __forceinline__ v8f wmma_bf(v16bf a, v16bf b, v8f c) {
  return __builtin_amdgcn_wmma_f32_16x16x32_bf16(false, a, false, b, (short)0, c,
                                                 RA, false);
}

// A-matrix 16x32 bf16 tile, contiguous K: lane<16 -> M=lane, K=0..15;
// lane>=16 -> M=lane-16, K=16..31. p already points at (rowM, kBase).
__device__ __forceinline__ v16bf load_a_tile(const bf16_t* p) {
  v8bf lo = *(const v8bf*)(p);
  v8bf hi = *(const v8bf*)(p + 8);
  v16bf r;
#pragma unroll
  for (int i = 0; i < 8; ++i) { r[i] = lo[i]; r[8 + i] = hi[i]; }
  return r;
}

// ---------------------------------------------------------------------------
// fp32 -> bf16 grid-stride convert
// ---------------------------------------------------------------------------
__global__ __launch_bounds__(256) void f2bf_kernel(const float* __restrict__ in,
                                                   bf16_t* __restrict__ out, int n) {
  for (int i = blockIdx.x * blockDim.x + threadIdx.x; i < n; i += gridDim.x * blockDim.x)
    out[i] = (bf16_t)in[i];
}

// ---------------------------------------------------------------------------
// init: dec = tanh(latent @ fc_w^T + fc_b)  (256 x 2048, K=256)
// dec in natural layout == initial fc_in. Also scatter h0 (bf16) and c0 (f32).
// grid = 512 blocks x 128 thr: mTile = blk>>5 (16), nTile = (blk&31)*4+wave (128)
// ---------------------------------------------------------------------------
__global__ __launch_bounds__(128) void init_fc_kernel(
    const bf16_t* __restrict__ lat, const bf16_t* __restrict__ fcw,
    const float* __restrict__ fcb, bf16_t* __restrict__ fcin,
    bf16_t* __restrict__ h0, float* __restrict__ c0) {
  const int lane = threadIdx.x & 31, wave = threadIdx.x >> 5;
  const int m0 = (blockIdx.x >> 5) * 16;
  const int n0 = (((blockIdx.x & 31) << 2) + wave) * 16;
  const int aRow = m0 + (lane & 15), aK = (lane >> 4) * 16;
  const int bRow = n0 + (lane & 15), bK = (lane >> 4) * 16;
  const bf16_t* aB = lat + (size_t)aRow * LATD + aK;
  const bf16_t* bB = fcw + (size_t)bRow * LATD + bK;
  v8f acc = {};
#pragma unroll
  for (int k = 0; k < LATD; k += 32)
    acc = wmma_bf<false>(load_a_tile(aB + k), *(const v16bf*)(bB + k), acc);

  const int n = n0 + (lane & 15);
  const float bias = fcb[n];
  const int mBase = m0 + (lane >> 4) * 8;
#pragma unroll
  for (int v = 0; v < 8; ++v) {
    const int i = mBase + v;
    float val = fast_tanh(acc[v] + bias);
    fcin[(size_t)i * FCW + n] = (bf16_t)val;
    // dec.reshape(2,256,1024): l = i/128, b = (i%128)*2 + (n/1024), k1 = n%1024
    const int l = i >> 7, bb = ((i & 127) << 1) + (n >> 10), k1 = n & 1023;
    if (k1 < HID) h0[((size_t)l * BSZ + bb) * HID + k1] = (bf16_t)val;
    else          c0[((size_t)l * BSZ + bb) * HID + (k1 - HID)] = val;
  }
}

// ---------------------------------------------------------------------------
// map: x = sigmoid(fc_in @ map_w^T + map_b)  (256 x 512, K=2048)
// Each wave owns TWO 16x16 N-tiles per A-tile (1 A load : 2 WMMAs, reuse-A).
// grid = 64 blocks x 128 thr: mTile = blk>>2 (16), nPair = (blk&3)*4+wave (16).
// ---------------------------------------------------------------------------
__global__ __launch_bounds__(128) void map_kernel(
    const bf16_t* __restrict__ fcin, const bf16_t* __restrict__ mapw,
    const float* __restrict__ mapb, float* __restrict__ outf,
    bf16_t* __restrict__ x0) {
  const int lane = threadIdx.x & 31, wave = threadIdx.x >> 5;
  const int m0 = (blockIdx.x >> 2) * 16;
  const int n0 = (((blockIdx.x & 3) << 2) + wave) * 32;   // pair of N-tiles
  const int aRow = m0 + (lane & 15), aK = (lane >> 4) * 16;
  const int bRow = n0 + (lane & 15), bK = (lane >> 4) * 16;
  const bf16_t* aB  = fcin + (size_t)aRow * FCW + aK;
  const bf16_t* bB0 = mapw + (size_t)bRow * FCW + bK;
  const bf16_t* bB1 = mapw + (size_t)(bRow + 16) * FCW + bK;
  v8f acc0 = {}, acc1 = {};
#pragma unroll 4
  for (int k = 0; k < FCW; k += 32) {
    v16bf a = load_a_tile(aB + k);
    acc0 = wmma_bf<true >(a, *(const v16bf*)(bB0 + k), acc0);
    acc1 = wmma_bf<false>(a, *(const v16bf*)(bB1 + k), acc1);
  }

  const int mBase = m0 + (lane >> 4) * 8;
#pragma unroll
  for (int t = 0; t < 2; ++t) {
    const int n = n0 + t * 16 + (lane & 15);
    const float bias = mapb[n];
    v8f& acc = t ? acc1 : acc0;
#pragma unroll
    for (int v = 0; v < 8; ++v) {
      const int b = mBase + v;
      float s = sigm(acc[v] + bias);
      outf[(size_t)b * HID + n] = s;
      x0[(size_t)b * HID + n]   = (bf16_t)s;
    }
  }
}

// ---------------------------------------------------------------------------
// LSTM cell: g = x@wih^T + h@whh^T + bih + bhh ; gates i,f,g,o (512 each).
// One wave owns a 16x16 tile of all four gates (4 accumulators, reuse-A
// across the gate quartet). grid = 128 blocks x 128 thr.
// Writes: c (tanh(c2), f32, in place), hnext (tanh(h2), bf16, ping-pong),
// xnext (h2 pre-tanh, bf16, layer-1 input; null for layer 1), fc_in scatter.
// ---------------------------------------------------------------------------
__global__ __launch_bounds__(128) void lstm_kernel(
    const bf16_t* __restrict__ x, const bf16_t* __restrict__ hprev,
    const bf16_t* __restrict__ wih, const bf16_t* __restrict__ whh,
    const float* __restrict__ bih, const float* __restrict__ bhh,
    float* __restrict__ cbuf, bf16_t* __restrict__ hnext,
    bf16_t* __restrict__ xnext, bf16_t* __restrict__ fcin, int layer) {
  const int lane = threadIdx.x & 31, wave = threadIdx.x >> 5;
  const int m0 = (blockIdx.x >> 3) * 16;
  const int n0 = (((blockIdx.x & 7) << 2) + wave) * 16;
  const int aRow = m0 + (lane & 15), aK = (lane >> 4) * 16;
  const int bRow = n0 + (lane & 15), bK = (lane >> 4) * 16;

  v8f aI = {}, aF = {}, aG = {}, aO = {};

  // x part: K = 512 over wih
  {
    const bf16_t* aB = x + (size_t)aRow * HID + aK;
    const bf16_t* bI = wih + ((size_t)(0 * HID + bRow)) * HID + bK;
    const bf16_t* bF = wih + ((size_t)(1 * HID + bRow)) * HID + bK;
    const bf16_t* bG = wih + ((size_t)(2 * HID + bRow)) * HID + bK;
    const bf16_t* bO = wih + ((size_t)(3 * HID + bRow)) * HID + bK;
#pragma unroll 2
    for (int k = 0; k < HID; k += 32) {
      v16bf a = load_a_tile(aB + k);
      aI = wmma_bf<true >(a, *(const v16bf*)(bI + k), aI);
      aF = wmma_bf<true >(a, *(const v16bf*)(bF + k), aF);
      aG = wmma_bf<true >(a, *(const v16bf*)(bG + k), aG);
      aO = wmma_bf<false>(a, *(const v16bf*)(bO + k), aO);
    }
  }
  // h part: K = 512 over whh
  {
    const bf16_t* aB = hprev + (size_t)aRow * HID + aK;
    const bf16_t* bI = whh + ((size_t)(0 * HID + bRow)) * HID + bK;
    const bf16_t* bF = whh + ((size_t)(1 * HID + bRow)) * HID + bK;
    const bf16_t* bG = whh + ((size_t)(2 * HID + bRow)) * HID + bK;
    const bf16_t* bO = whh + ((size_t)(3 * HID + bRow)) * HID + bK;
#pragma unroll 2
    for (int k = 0; k < HID; k += 32) {
      v16bf a = load_a_tile(aB + k);
      aI = wmma_bf<true >(a, *(const v16bf*)(bI + k), aI);
      aF = wmma_bf<true >(a, *(const v16bf*)(bF + k), aF);
      aG = wmma_bf<true >(a, *(const v16bf*)(bG + k), aG);
      aO = wmma_bf<false>(a, *(const v16bf*)(bO + k), aO);
    }
  }

  const int n = n0 + (lane & 15);
  const float biasI = bih[0 * HID + n] + bhh[0 * HID + n];
  const float biasF = bih[1 * HID + n] + bhh[1 * HID + n];
  const float biasG = bih[2 * HID + n] + bhh[2 * HID + n];
  const float biasO = bih[3 * HID + n] + bhh[3 * HID + n];
  const int mBase = m0 + (lane >> 4) * 8;
#pragma unroll
  for (int v = 0; v < 8; ++v) {
    const int b = mBase + v;
    float gi = sigm(aI[v] + biasI);
    float gf = sigm(aF[v] + biasF);
    float gg = fast_tanh(aG[v] + biasG);
    float go = sigm(aO[v] + biasO);
    float cold = cbuf[(size_t)b * HID + n];
    float c2 = gf * cold + gi * gg;
    float h2 = go * fast_tanh(c2);
    float th = fast_tanh(h2);
    float tc = fast_tanh(c2);
    cbuf[(size_t)b * HID + n]  = tc;           // carried c = tanh(c2)
    hnext[(size_t)b * HID + n] = (bf16_t)th;   // carried h = tanh(h2)
    if (xnext) xnext[(size_t)b * HID + n] = (bf16_t)h2;  // layer-1 input (pre-tanh)
    // fc_in scatter: row = l*128 + b/2, col = (b&1)*1024 + {0,512} + n
    const size_t row = (size_t)layer * 128 + (b >> 1);
    const size_t col = (size_t)(b & 1) * 1024 + n;
    fcin[row * FCW + col]       = (bf16_t)th;
    fcin[row * FCW + col + HID] = (bf16_t)tc;
  }
}

// ---------------------------------------------------------------------------
extern "C" void kernel_launch(void* const* d_in, const int* in_sizes, int n_in,
                              void* d_out, int out_size, void* d_ws, size_t ws_size,
                              hipStream_t stream) {
  const float* latent = (const float*)d_in[0];
  const float* fc_w   = (const float*)d_in[1];
  const float* fc_b   = (const float*)d_in[2];
  const float* map_w  = (const float*)d_in[3];
  const float* map_b  = (const float*)d_in[4];
  const float* wih0   = (const float*)d_in[5];
  const float* whh0   = (const float*)d_in[6];
  const float* bih0   = (const float*)d_in[7];
  const float* bhh0   = (const float*)d_in[8];
  const float* wih1   = (const float*)d_in[9];
  const float* whh1   = (const float*)d_in[10];
  const float* bih1   = (const float*)d_in[11];
  const float* bhh1   = (const float*)d_in[12];
  (void)in_sizes; (void)n_in; (void)out_size; (void)ws_size;

  // ---- workspace carve-up (bf16 element counts) ----
  bf16_t* lat_bf  = (bf16_t*)d_ws;                       // 256*256
  bf16_t* fcw_bf  = lat_bf  + BSZ * LATD;                // 2048*256
  bf16_t* mapw_bf = fcw_bf  + FCW * LATD;                // 512*2048
  bf16_t* wih0_bf = mapw_bf + HID * FCW;                 // 2048*512 each
  bf16_t* whh0_bf = wih0_bf + 4 * HID * HID;
  bf16_t* wih1_bf = whh0_bf + 4 * HID * HID;
  bf16_t* whh1_bf = wih1_bf + 4 * HID * HID;
  bf16_t* fcin    = whh1_bf + 4 * HID * HID;             // 256*2048
  bf16_t* hbufA   = fcin    + BSZ * FCW;                 // 2*256*512 (both layers)
  bf16_t* hbufB   = hbufA   + 2 * BSZ * HID;
  bf16_t* x0      = hbufB   + 2 * BSZ * HID;             // 256*512
  bf16_t* x1      = x0      + BSZ * HID;
  float*  cbuf    = (float*)(x1 + BSZ * HID);            // 2*256*512 f32

  // ---- one-time bf16 conversions ----
  auto conv = [&](const float* src, bf16_t* dst, int n) {
    f2bf_kernel<<<(n + 1023) / 1024, 256, 0, stream>>>(src, dst, n);
  };
  conv(latent, lat_bf,  BSZ * LATD);
  conv(fc_w,   fcw_bf,  FCW * LATD);
  conv(map_w,  mapw_bf, HID * FCW);
  conv(wih0,   wih0_bf, 4 * HID * HID);
  conv(whh0,   whh0_bf, 4 * HID * HID);
  conv(wih1,   wih1_bf, 4 * HID * HID);
  conv(whh1,   whh1_bf, 4 * HID * HID);

  // ---- initial state: tanh(latent @ fc_w^T + fc_b) -> fcin, hbufA, cbuf ----
  init_fc_kernel<<<512, 128, 0, stream>>>(lat_bf, fcw_bf, fc_b, fcin, hbufA, cbuf);

  // ---- 256 sequential steps ----
  bf16_t* hb[2] = {hbufA, hbufB};
  for (int s = 0; s < SEQ; ++s) {
    float* step_out = (float*)d_out + (size_t)s * BSZ * HID;
    map_kernel<<<64, 128, 0, stream>>>(fcin, mapw_bf, map_b, step_out, x0);

    bf16_t* hprev = hb[s & 1];
    bf16_t* hnext = hb[(s + 1) & 1];
    lstm_kernel<<<128, 128, 0, stream>>>(x0, hprev, wih0_bf, whh0_bf, bih0, bhh0,
                                         cbuf, hnext, x1, fcin, 0);
    lstm_kernel<<<128, 128, 0, stream>>>(x1, hprev + (size_t)BSZ * HID,
                                         wih1_bf, whh1_bf, bih1, bhh1,
                                         cbuf + (size_t)BSZ * HID,
                                         hnext + (size_t)BSZ * HID,
                                         (bf16_t*)nullptr, fcin, 1);
  }
}
